// Attention_18820546691177
// MI455X (gfx1250) — matrix operand order repaired and checked by reading, not compiled, not run on hardware
//
#include <hip/hip_runtime.h>

#define DEV static __device__ __forceinline__

typedef __attribute__((ext_vector_type(16))) __bf16 v16bf;
typedef __attribute__((ext_vector_type(8)))  __bf16 v8bf;
typedef __attribute__((ext_vector_type(8)))  float  v8f;

constexpr int Bn = 4, Tn = 2048, Dn = 1024, Hn = 16, HDn = 64;

DEV __bf16 f2bf(float f) {
  union { float f; unsigned u; } in; in.f = f;
  unsigned u = in.u;
  unsigned r = u + 0x7FFFu + ((u >> 16) & 1u);   // round-to-nearest-even
  union { unsigned short s; __bf16 b; } out; out.s = (unsigned short)(r >> 16);
  return out.b;
}

// A operand: 16x32 bf16, M = lane&15, K-chunks at k0+8*hi and k0+16+8*hi (hi = lane>>4)
DEV v16bf load_a_frag(const __bf16* base, int ld, int m0, int k0) {
  const int lane = threadIdx.x & 31;
  const int m  = m0 + (lane & 15);
  const int kh = (lane >> 4) * 8;
  const __bf16* p = base + (size_t)m * ld + (k0 + kh);
  v16bf a;
  ((v8bf*)&a)[0] = *(const v8bf*)(p);
  ((v8bf*)&a)[1] = *(const v8bf*)(p + 16);
  return a;
}

// B operand: 32x16 bf16 loaded from B^T (rows of bt are columns of B).
// Column N = lane&15; K-chunks at k0+8*hi and k0+16+8*hi (hi = lane>>4).
DEV v16bf load_bt_frag(const __bf16* bt, int ld, int n0, int k0) {
  const int lane = threadIdx.x & 31;
  const int n  = n0 + (lane & 15);
  const int kh = (lane >> 4) * 8;
  const __bf16* p = bt + (size_t)n * ld + (k0 + kh);
  v16bf b;
  ((v8bf*)&b)[0] = *(const v8bf*)(p);
  ((v8bf*)&b)[1] = *(const v8bf*)(p + 16);
  return b;
}

DEV v8f wmma_bf16(v16bf a, v16bf b, v8f c) {
  return __builtin_amdgcn_wmma_f32_16x16x32_bf16(false, a, false, b, (short)0, c,
                                                 false, false);
}

// ---- gfx1250 async global->LDS copy (ASYNCcnt-tracked), inline asm ---------
typedef __attribute__((address_space(3))) const char as3_char;

DEV unsigned lds_byte_off(const void* p) {
  return (unsigned)(unsigned long long)(as3_char*)p;   // generic -> LDS offset
}

DEV void async_copy_b128(const void* gsrc, const void* ldst) {
  const unsigned l = lds_byte_off(ldst);
  const unsigned long long g = (unsigned long long)(size_t)gsrc;
  asm volatile("global_load_async_to_lds_b128 %0, %1, off"
               :
               : "v"(l), "v"(g)
               : "memory");
}

template <int N> DEV void wait_asynccnt() {
  asm volatile("s_wait_asynccnt %0" : : "n"(N) : "memory");
}

// Cooperative prefetch of one 32-key K tile (32x64, contiguous 4KB) and one
// V^T tile (64 dims x 32 keys, 64B segments) into LDS. 4 async b128 per thread.
DEV void prefetch_tiles(const __bf16* kp, const __bf16* vp, int j0,
                        __bf16* kbuf, __bf16* vbuf, int tid) {
  const char* kg = (const char*)(kp + (size_t)j0 * HDn);   // 4KB contiguous
  char* kl = (char*)kbuf;
  async_copy_b128(kg + tid * 32,      kl + tid * 32);
  async_copy_b128(kg + tid * 32 + 16, kl + tid * 32 + 16);
#pragma unroll
  for (int i = 0; i < 2; ++i) {
    const int c = tid * 2 + i;          // 256 chunks of 16B
    const int d = c >> 2;               // dim row 0..63
    const int cofs = (c & 3) * 16;      // 4 chunks per 64B row
    const char* vg = (const char*)(vp + (size_t)d * Tn + j0) + cofs;
    char* vl = (char*)(vbuf + d * 32) + cofs;
    async_copy_b128(vg, vl);
  }
}

// ---------------------------------------------------------------- convert x
__global__ void convert_x_kernel(const float* __restrict__ x,
                                 __bf16* __restrict__ xb, int n) {
  for (int i = blockIdx.x * blockDim.x + threadIdx.x; i < n;
       i += gridDim.x * blockDim.x)
    xb[i] = f2bf(x[i]);
}

// ---------------------------------------------------- transpose W -> bf16 W^T
__global__ void transpose_w_kernel(const float* __restrict__ wq,
                                   const float* __restrict__ wk,
                                   const float* __restrict__ wv,
                                   __bf16* __restrict__ tq,
                                   __bf16* __restrict__ tk,
                                   __bf16* __restrict__ tv) {
  const int idx = blockIdx.x * blockDim.x + threadIdx.x;   // 0 .. D*D-1
  const float* w = (blockIdx.y == 0) ? wq : (blockIdx.y == 1) ? wk : wv;
  __bf16*      t = (blockIdx.y == 0) ? tq : (blockIdx.y == 1) ? tk : tv;
  const int n = idx >> 10;        // output row   (original column)
  const int k = idx & (Dn - 1);   // output col   (original row)
  t[idx] = f2bf(w[k * Dn + n]);
}

// ------------------------------------------------------------- fused QKV GEMM
// Each wave: 16 rows x 64 cols (= one head's HD) of one of Q/K/V.
// Q written scaled by 1/sqrt(HD); V written transposed [B,H,HD,T].
__global__ __launch_bounds__(128) void qkv_gemm_kernel(
    const __bf16* __restrict__ xb,
    const __bf16* __restrict__ wtq, const __bf16* __restrict__ wtk,
    const __bf16* __restrict__ wtv,
    const float* __restrict__ bq, const float* __restrict__ bk,
    const float* __restrict__ bv,
    __bf16* __restrict__ Qh, __bf16* __restrict__ Kh, __bf16* __restrict__ Vt) {
  const int wid  = threadIdx.x >> 5;
  const int lane = threadIdx.x & 31;
  const int mat  = blockIdx.z;                 // 0:q 1:k 2:v
  const int h    = blockIdx.y;
  const int m0   = (blockIdx.x * 4 + wid) * 16;
  const int n0   = h * HDn;

  const __bf16* wt   = (mat == 0) ? wtq : (mat == 1) ? wtk : wtv;
  const float*  bias = (mat == 0) ? bq  : (mat == 1) ? bk  : bv;
  const float   scale = (mat == 0) ? 0.125f : 1.0f;   // 1/sqrt(64) folded into Q

  v8f acc[4];
#pragma unroll
  for (int s = 0; s < 4; ++s)
#pragma unroll
    for (int r = 0; r < 8; ++r) acc[s][r] = 0.0f;

  for (int k0 = 0; k0 < Dn; k0 += 32) {
    v16bf a = load_a_frag(xb, Dn, m0, k0);
#pragma unroll
    for (int s = 0; s < 4; ++s) {
      v16bf bm = load_bt_frag(wt, Dn, n0 + s * 16, k0);
      acc[s] = wmma_bf16(a, bm, acc[s]);
    }
  }

  const int col = lane & 15;
  const int hi  = lane >> 4;
#pragma unroll
  for (int s = 0; s < 4; ++s) {
#pragma unroll
    for (int r = 0; r < 8; ++r) {
      const int m = m0 + r + 8 * hi;
      const int d = s * 16 + col;
      const float v = (acc[s][r] + bias[n0 + d]) * scale;
      const int b = m >> 11;          // m / T
      const int t = m & (Tn - 1);     // m % T
      const size_t bh = (size_t)(b * Hn + h);
      if (mat == 0)      Qh[(bh * Tn + t) * HDn + d] = f2bf(v);
      else if (mat == 1) Kh[(bh * Tn + t) * HDn + d] = f2bf(v);
      else               Vt[(bh * HDn + d) * Tn + t] = f2bf(v);
    }
  }
}

// --------------------------------------------------------------- flash attn
// One block = 4 waves = 4 query tiles of one (b,h). K/V 32-key tiles are
// double-buffered in LDS via async global->LDS DMA, shared by all 4 waves.
__global__ __launch_bounds__(128) void flash_attn_kernel(
    const __bf16* __restrict__ Qh, const __bf16* __restrict__ Kh,
    const __bf16* __restrict__ Vt, float* __restrict__ out) {
  __shared__ __align__(16) __bf16 Klds[2][32 * 64];   // 2 x 4KB
  __shared__ __align__(16) __bf16 Vlds[2][64 * 32];   // 2 x 4KB
  __shared__ __align__(16) __bf16 Plds[4][16 * 32];   // per-wave P tile

  const int tid  = threadIdx.x;
  const int wid  = tid >> 5;
  const int lane = tid & 31;
  const int b    = blockIdx.z;
  const int h    = blockIdx.y;
  const int t0   = (blockIdx.x * 4 + wid) * 16;

  const size_t bh = (size_t)(b * Hn + h);
  const __bf16* qp = Qh + bh * Tn * HDn;
  const __bf16* kp = Kh + bh * Tn * HDn;
  const __bf16* vp = Vt + bh * HDn * Tn;

  const v16bf aq0 = load_a_frag(qp, HDn, t0, 0);
  const v16bf aq1 = load_a_frag(qp, HDn, t0, 32);

  v8f o[4];
  float mrow[8], lrow[8];
#pragma unroll
  for (int s = 0; s < 4; ++s)
#pragma unroll
    for (int r = 0; r < 8; ++r) o[s][r] = 0.0f;
#pragma unroll
  for (int r = 0; r < 8; ++r) { mrow[r] = -3.0e38f; lrow[r] = 0.0f; }

  const int col = lane & 15;
  const int hi  = lane >> 4;
  __bf16* pl = &Plds[wid][0];

  prefetch_tiles(kp, vp, 0, Klds[0], Vlds[0], tid);   // prologue prefetch

  for (int j0 = 0; j0 < Tn; j0 += 32) {
    const int buf = (j0 >> 5) & 1;
    if (j0 + 32 < Tn) {
      prefetch_tiles(kp, vp, j0 + 32, Klds[buf ^ 1], Vlds[buf ^ 1], tid);
      wait_asynccnt<4>();    // my 4 next-tile ops may remain; current tile done
    } else {
      wait_asynccnt<0>();
    }
    __syncthreads();         // current K/V tile visible to all waves

    const __bf16* kt = &Klds[buf][0];
    const __bf16* vt = &Vlds[buf][0];

    v8f s0, s1;
#pragma unroll
    for (int r = 0; r < 8; ++r) { s0[r] = 0.0f; s1[r] = 0.0f; }

    {  // S = Q * K^T for local keys 0..31 (ds_load_b128 from LDS tile)
      v16bf bk0 = load_bt_frag(kt, HDn, 0, 0);
      s0 = wmma_bf16(aq0, bk0, s0);
      v16bf bk1 = load_bt_frag(kt, HDn, 0, 32);
      s0 = wmma_bf16(aq1, bk1, s0);
      v16bf bk2 = load_bt_frag(kt, HDn, 16, 0);
      s1 = wmma_bf16(aq0, bk2, s1);
      v16bf bk3 = load_bt_frag(kt, HDn, 16, 32);
      s1 = wmma_bf16(aq1, bk3, s1);
    }

    float alpha[8];
#pragma unroll
    for (int r = 0; r < 8; ++r) {
      const float x0 = s0[r], x1 = s1[r];
      float mx0 = x0, mx1 = x1;
#pragma unroll
      for (int mk = 1; mk <= 8; mk <<= 1) {   // reduce within 16-lane half
        mx0 = fmaxf(mx0, __shfl_xor(mx0, mk, 32));
        mx1 = fmaxf(mx1, __shfl_xor(mx1, mk, 32));
      }
      const float mnew = fmaxf(mrow[r], fmaxf(mx0, mx1));
      const float al = __expf(mrow[r] - mnew);
      mrow[r] = mnew;
      alpha[r] = al;
      const float p0 = __expf(x0 - mnew);
      const float p1 = __expf(x1 - mnew);
      float rs = p0 + p1;
#pragma unroll
      for (int mk = 1; mk <= 8; mk <<= 1) rs += __shfl_xor(rs, mk, 32);
      lrow[r] = lrow[r] * al + rs;
      const int row = r + 8 * hi;
      pl[row * 32 + col]      = f2bf(p0);     // C-layout -> LDS tile
      pl[row * 32 + 16 + col] = f2bf(p1);
    }

    __syncthreads();                          // fence P stores
    const v16bf ap = load_a_frag(pl, 32, 0, 0);  // P as 16x32 A operand

#pragma unroll
    for (int s = 0; s < 4; ++s) {
#pragma unroll
      for (int r = 0; r < 8; ++r) o[s][r] *= alpha[r];   // online rescale
      v16bf bvf = load_bt_frag(vt, 32, s * 16, 0);       // V^T tile from LDS
      o[s] = wmma_bf16(ap, bvf, o[s]);
    }

    __syncthreads();   // all reads of this buffer done before it is re-filled
  }

#pragma unroll
  for (int s = 0; s < 4; ++s)
#pragma unroll
    for (int r = 0; r < 8; ++r) {
      const int t    = t0 + r + 8 * hi;
      const int dcol = h * HDn + s * 16 + col;
      out[((size_t)b * Tn + t) * Dn + dcol] = o[s][r] / lrow[r];
    }
}

// ---------------------------------------------------------------------------
extern "C" void kernel_launch(void* const* d_in, const int* in_sizes, int n_in,
                              void* d_out, int out_size, void* d_ws,
                              size_t ws_size, hipStream_t stream) {
  const float* x  = (const float*)d_in[0];
  const float* Wq = (const float*)d_in[1];
  const float* bq = (const float*)d_in[2];
  const float* Wk = (const float*)d_in[3];
  const float* bk = (const float*)d_in[4];
  const float* Wv = (const float*)d_in[5];
  const float* bv = (const float*)d_in[6];
  float* out = (float*)d_out;

  char* ws = (char*)d_ws;
  size_t off = 0;
  auto take = [&](size_t bytes) -> char* {
    char* p = ws + off;
    off += (bytes + 255) & ~(size_t)255;
    return p;
  };
  __bf16* xb  = (__bf16*)take((size_t)Bn * Tn * Dn * 2);  // 16 MiB
  __bf16* wtq = (__bf16*)take((size_t)Dn * Dn * 2);       //  2 MiB
  __bf16* wtk = (__bf16*)take((size_t)Dn * Dn * 2);
  __bf16* wtv = (__bf16*)take((size_t)Dn * Dn * 2);
  __bf16* Qh  = (__bf16*)take((size_t)Bn * Tn * Dn * 2);  // 16 MiB each
  __bf16* Kh  = (__bf16*)take((size_t)Bn * Tn * Dn * 2);
  __bf16* Vt  = (__bf16*)take((size_t)Bn * Tn * Dn * 2);

  convert_x_kernel<<<2048, 256, 0, stream>>>(x, xb, Bn * Tn * Dn);
  transpose_w_kernel<<<dim3(Dn * Dn / 256, 3), 256, 0, stream>>>(
      Wq, Wk, Wv, wtq, wtk, wtv);
  qkv_gemm_kernel<<<dim3((Bn * Tn) / 64, Hn, 3), 128, 0, stream>>>(
      xb, wtq, wtk, wtv, bq, bk, bv, Qh, Kh, Vt);
  flash_attn_kernel<<<dim3(Tn / 64, Hn, Bn), 128, 0, stream>>>(Qh, Kh, Vt, out);
}
